// SGRUCell_86492051407610
// MI455X (gfx1250) — matrix operand order repaired and checked by reading, not compiled, not run on hardware
//
#include <hip/hip_runtime.h>

// SGRUCell for MI455X (gfx1250):
//   - one-time f32->bf16 conversion/transpose prep kernels
//   - bf16 WMMA GEMMs (fp32 accumulate) fed by double-buffered
//     global_load_async_to_lds_b128 (ASYNCcnt pipelined)
//   - fp32 epilogues: kernel_c gather + rank-2 'pre', biases, activations
//   - softmax mixture head in fp32
//
// B=4096, U=1024, C=5000, M=20
#define BATCH   4096
#define UDIM    1024
#define FOURU   4096
#define SPITCH  1026   // states row pitch (U+2)
#define OPITCH  1086   // out row pitch (U + 3M + 2)
#define NSPITCH 1026   // new_state row pitch (U+2)
#define MGMM    20

typedef __attribute__((ext_vector_type(16))) __bf16 v16bf;
typedef __attribute__((ext_vector_type(8)))  float  v8f;

__device__ __forceinline__ float sigmoidf_(float x) { return 1.0f / (1.0f + expf(-x)); }

// One async B128 transfer per lane: 16 bytes global -> LDS, tracked by ASYNCcnt.
__device__ __forceinline__ void async_b128(unsigned lds_off, const void* g) {
  asm volatile("global_load_async_to_lds_b128 %0, %1, off"
               :: "v"(lds_off), "v"((unsigned long long)(size_t)g)
               : "memory");
}
#define WAIT_ASYNC(n) asm volatile("s_wait_asynccnt " #n ::: "memory")

// ---------------------------------------------------------------------------
// Prep kernels (one-time conversions; memory-bound, ~48 MB total traffic)
// ---------------------------------------------------------------------------

// rkT[n][k] = bf16(rk[k][n])  : 1024x4096 -> 4096x1024, LDS-tiled transpose
__global__ __launch_bounds__(256) void transpose_rk(
    const float* __restrict__ src, __bf16* __restrict__ dst)
{
  __shared__ float t[32][33];
  const int tx = threadIdx.x & 31, ty = threadIdx.x >> 5;
  const int n0 = blockIdx.x * 32, k0 = blockIdx.y * 32;
  #pragma unroll
  for (int i = 0; i < 4; ++i)
    t[ty + i * 8][tx] = src[(size_t)(k0 + ty + i * 8) * FOURU + n0 + tx];
  __syncthreads();
  #pragma unroll
  for (int i = 0; i < 4; ++i)
    dst[(size_t)(n0 + ty + i * 8) * UDIM + k0 + tx] = (__bf16)t[tx][ty + i * 8];
}

// wgT[n][k] = bf16(wg[k][n]) for n<60, else 0   (64x1024)
__global__ __launch_bounds__(256) void make_wgT(
    const float* __restrict__ wg, __bf16* __restrict__ wgT)
{
  int id = blockIdx.x * 256 + threadIdx.x;  // 64*1024
  int n = id >> 10, k = id & 1023;
  float v = (n < 3 * MGMM) ? wg[(size_t)k * (3 * MGMM) + n] : 0.0f;
  wgT[id] = (__bf16)v;
}

// xh[b][c] = bf16(states[b][c])  (contiguous bf16 copy of h_tm1)
__global__ __launch_bounds__(256) void conv_h(
    const float* __restrict__ states, __bf16* __restrict__ xh)
{
  int id = blockIdx.x * 256 + threadIdx.x;  // B*U
  int r = id >> 10, c = id & 1023;
  xh[id] = (__bf16)states[(size_t)r * SPITCH + c];
}

// ---------------------------------------------------------------------------
// Epilogues (all fp32 math)
// ---------------------------------------------------------------------------

// pre(b, col) = kernel_c[ch[b], col] + d0*kernel_d[0,col] + d1*kernel_d[1,col]
struct PreCtx {
  const int*   ch;
  const float* states;
  const float* kc;
  const float* kd;
  __device__ __forceinline__ float pre(int row, int col) const {
    int   c  = ch[row];
    float d0 = states[(size_t)row * SPITCH + UDIM];
    float d1 = states[(size_t)row * SPITCH + UDIM + 1];
    return kc[(size_t)c * FOURU + col] + d0 * kd[col] + d1 * kd[FOURU + col];
  }
};

struct EpiZR {  // cols [0,2U): z=sigmoid(.), r=sigmoid(.); store z (f32), r*h (bf16)
  PreCtx p; const float* bias_z; const float* bias; float* z; __bf16* rh;
  __device__ void operator()(int row, int col, float s) const {
    float v = s + p.pre(row, col);
    if (col < UDIM) {
      z[(size_t)row * UDIM + col] = sigmoidf_(v + bias_z[col]);
    } else {
      int   c2 = col - UDIM;
      float r  = sigmoidf_(v + bias[c2]);
      rh[(size_t)row * UDIM + c2] = (__bf16)(r * p.states[(size_t)row * SPITCH + c2]);
    }
  }
};

struct EpiHH {  // cols [2U,3U): hh=tanh(.); h = z*h_tm1+(1-z)*hh -> bf16 + new_state f32
  PreCtx p; const float* bias; const float* z; __bf16* hb; float* ns;
  __device__ void operator()(int row, int col, float s) const {
    int   c2 = col - 2 * UDIM;
    float hh = tanhf(s + p.pre(row, col) + bias[UDIM + c2]);
    float zz = z[(size_t)row * UDIM + c2];
    float ht = p.states[(size_t)row * SPITCH + c2];
    float h  = zz * ht + (1.0f - zz) * hh;
    hb[(size_t)row * UDIM + c2]    = (__bf16)h;
    ns[(size_t)row * NSPITCH + c2] = h;
  }
};

struct EpiO {   // cols [3U,4U): o=tanh(.) -> out f32 + bf16 copy for GMM GEMM
  PreCtx p; const float* bias; __bf16* ob; float* out;
  __device__ void operator()(int row, int col, float s) const {
    int   c2 = col - 3 * UDIM;
    float o  = tanhf(s + p.pre(row, col) + bias[2 * UDIM + c2]);
    ob[(size_t)row * UDIM + c2]    = (__bf16)o;
    out[(size_t)row * OPITCH + c2] = o;
  }
};

struct EpiGMM { // gmm = o @ Wgmm + bgmm (cols<60 guarded by Nreal in kernel)
  const float* bg; float* gmm;
  __device__ void operator()(int row, int col, float s) const {
    gmm[(size_t)row * 64 + col] = s + bg[col];
  }
};

// ---------------------------------------------------------------------------
// Async-pipelined bf16 WMMA GEMM
//   X : [B][1024] bf16 (contiguous)
//   Wt: [Ntot][1024] bf16 (pre-transposed: row n = output column n)
//   Block tile 32(M) x 64(N), 8 waves of one 16x16 WMMA tile each,
//   K staged 32/chunk through double-buffered LDS via async B128.
// ---------------------------------------------------------------------------
template <typename Epi>
__global__ __launch_bounds__(256) void gemm_async(
    const __bf16* __restrict__ Xb, const __bf16* __restrict__ Wt,
    int wrow0, int Nreal, Epi epi)
{
  __shared__ __attribute__((aligned(16))) __bf16 Xs[2][32][40];
  __shared__ __attribute__((aligned(16))) __bf16 Ws[2][64][40];

  const int tid  = threadIdx.x;
  const int lane = tid & 31;
  const int w    = tid >> 5;      // wave 0..7
  const int wm   = w >> 2;        // M sub-tile
  const int wn   = w & 3;         // N sub-tile
  const int rowBase = blockIdx.y * 32;
  const int colBase = blockIdx.x * 64;
  const int hf  = lane >> 4;
  const int l15 = lane & 15;

  // Async copy assignments (16B per lane):
  //  X tile (32 rows x 64B): threads 0..127, waves 0-3  -> 1 instr/wave extra
  //  W tile (64 rows x 64B): all 256 threads            -> 1 instr/wave
  const int xr = tid >> 2;            // X row (valid for tid<128)
  const int xc = (tid & 3) * 8;       // bf16 offset within row
  const int wr = tid >> 2;            // W row 0..63
  const int wc = (tid & 3) * 8;
  const __bf16* xsrc = Xb + (size_t)(rowBase + xr) * UDIM + xc;
  const __bf16* wsrc = Wt + (size_t)(wrow0 + colBase + wr) * UDIM + wc;

  auto issue = [&](int k0, int buf) {
    if (tid < 128)
      async_b128((unsigned)(size_t)&Xs[buf][xr][xc], xsrc + k0);
    async_b128((unsigned)(size_t)&Ws[buf][wr][wc], wsrc + k0);
  };

  issue(0, 0);
  v8f acc = {};

  for (int kc = 0; kc < 32; ++kc) {
    const int buf = kc & 1;
    if (kc + 1 < 32) {
      issue((kc + 1) * 32, buf ^ 1);
      // In-order completion: waiting down to the just-issued count
      // guarantees chunk kc's transfers have landed.
      if (w < 4) WAIT_ASYNC(2);
      else       WAIT_ASYNC(1);
    } else {
      WAIT_ASYNC(0);
    }
    __syncthreads();

    // A fragment: contiguous K order matching B:
    //   lane<16: e0..15->K0..15; lane>=16: e0..15->K16..31.
    v16bf a, b;
    const int m = wm * 16 + l15;
    #pragma unroll
    for (int e = 0; e < 8; ++e) {
      a[e]     = Xs[buf][m][16 * hf + e];
      a[e + 8] = Xs[buf][m][16 * hf + 8 + e];
    }
    // B fragment: contiguous K order matching A: lanes 0-15 K=0..15, lanes 16-31 K=16..31.
    const int n = wn * 16 + l15;
    #pragma unroll
    for (int e = 0; e < 16; ++e) b[e] = Ws[buf][n][16 * hf + e];

    acc = __builtin_amdgcn_wmma_f32_16x16x32_bf16(
        false, a, false, b, (short)0, acc, false, false);
    __syncthreads();
  }

  // Epilogue. C/D layout: VGPR v -> M = v + 8*half, N = lane&15.
  const int lcol = colBase + wn * 16 + l15;
  if (lcol < Nreal) {
    const int col = wrow0 + lcol;
    #pragma unroll
    for (int v = 0; v < 8; ++v) {
      int row = rowBase + wm * 16 + 8 * hf + v;
      epi(row, col, acc[v]);
    }
  }
}

// ---------------------------------------------------------------------------
// Softmax over 20 mixture logits + weighted means; one thread per batch row.
// ---------------------------------------------------------------------------
__global__ __launch_bounds__(256) void gmm_finalize(
    const float* __restrict__ gmm, float* __restrict__ out, float* __restrict__ ns)
{
  int row = blockIdx.x * blockDim.x + threadIdx.x;
  if (row >= BATCH) return;
  const float* g = gmm + (size_t)row * 64;
  float e[MGMM];
  float sum = 0.0f;
  #pragma unroll
  for (int i = 0; i < MGMM; ++i) {
    float x = expf(g[i]);
    x = fminf(fmaxf(x, 1e-10f), 1e10f);   // exp_safe clip
    e[i] = x; sum += x;
  }
  float inv = 1.0f / sum;
  float xp = 0.0f, yp = 0.0f;
  float* orow = out + (size_t)row * OPITCH;
  #pragma unroll
  for (int i = 0; i < MGMM; ++i) {
    float pi  = e[i] * inv;
    float mux = g[MGMM + i];
    float muy = g[2 * MGMM + i];
    orow[UDIM + i]            = pi;
    orow[UDIM + MGMM + i]     = mux;
    orow[UDIM + 2 * MGMM + i] = muy;
    xp += pi * mux; yp += pi * muy;
  }
  orow[UDIM + 3 * MGMM]     = xp;
  orow[UDIM + 3 * MGMM + 1] = yp;
  float* nrow = ns + (size_t)row * NSPITCH;
  nrow[UDIM]     = xp;
  nrow[UDIM + 1] = yp;
}

// ---------------------------------------------------------------------------
extern "C" void kernel_launch(void* const* d_in, const int* in_sizes, int n_in,
                              void* d_out, int out_size, void* d_ws, size_t ws_size,
                              hipStream_t stream) {
  const int*   inputs = (const int*)  d_in[0];  // (B,1)
  const float* states = (const float*)d_in[1];  // (B,1026)
  const float* bias_z = (const float*)d_in[2];  // (1024,)
  const float* rk     = (const float*)d_in[3];  // (1024,4096)
  const float* kc     = (const float*)d_in[4];  // (5000,4096)
  const float* bias   = (const float*)d_in[5];  // (3072,)
  const float* kd     = (const float*)d_in[6];  // (2,4096)
  const float* wg     = (const float*)d_in[7];  // (1024,60)
  const float* bg     = (const float*)d_in[8];  // (60,)

  float* out = (float*)d_out;                   // (B,1086)
  float* ns  = out + (size_t)BATCH * OPITCH;    // (B,1026)

  // Workspace layout (~49 MB): rkT | wgT | xh | rh | h | z | gmm
  char* p = (char*)d_ws;
  __bf16* rkT = (__bf16*)p; p += (size_t)FOURU * UDIM * 2;   // 8 MB
  __bf16* wgT = (__bf16*)p; p += (size_t)64 * UDIM * 2;      // 128 KB
  __bf16* xh  = (__bf16*)p; p += (size_t)BATCH * UDIM * 2;   // 8 MB
  __bf16* rhb = (__bf16*)p; p += (size_t)BATCH * UDIM * 2;   // 8 MB
  __bf16* hb  = (__bf16*)p; p += (size_t)BATCH * UDIM * 2;   // 8 MB
  float*  z   = (float*)p;  p += (size_t)BATCH * UDIM * 4;   // 16 MB
  float*  gmm = (float*)p;  p += (size_t)BATCH * 64 * 4;     // 1 MB
  __bf16* ob  = rhb;   // r*h dead after K2: reuse its region for o (bf16)

  PreCtx pc{inputs, states, kc, kd};
  dim3 blk(256);

  // Prep: transpose/convert weights + h_tm1 to bf16
  transpose_rk<<<dim3(FOURU / 32, UDIM / 32), blk, 0, stream>>>(rk, rkT);
  make_wgT<<<dim3(64 * UDIM / 256), blk, 0, stream>>>(wg, wgT);
  conv_h<<<dim3(BATCH * UDIM / 256), blk, 0, stream>>>(states, xh);

  // K1: zr = h_tm1 @ RK[:, :2U] + pre + biases -> z, r*h_tm1
  gemm_async<<<dim3(2 * UDIM / 64, BATCH / 32), blk, 0, stream>>>(
      xh, rkT, 0, 2 * UDIM, EpiZR{pc, bias_z, bias, z, rhb});
  // K2: hh = tanh((r*h) @ RK[:, 2U:3U] + pre + bias); h = z*h_tm1 + (1-z)*hh
  gemm_async<<<dim3(UDIM / 64, BATCH / 32), blk, 0, stream>>>(
      rhb, rkT, 2 * UDIM, UDIM, EpiHH{pc, bias, z, hb, ns});
  // K3: o = tanh(h @ RK[:, 3U:] + pre + bias)
  gemm_async<<<dim3(UDIM / 64, BATCH / 32), blk, 0, stream>>>(
      hb, rkT, 3 * UDIM, UDIM, EpiO{pc, bias, ob, out});
  // K4: gmm = o @ Wgmm + bgmm (60 cols, W pre-padded to 64 rows)
  gemm_async<<<dim3(1, BATCH / 32), blk, 0, stream>>>(
      ob, wgT, 0, 3 * MGMM, EpiGMM{bg, gmm});
  // K5: softmax + predictions + final packing
  gmm_finalize<<<dim3(BATCH / 256), dim3(256), 0, stream>>>(gmm, out, ns);
}
